// RefineIUGraphLayer_87514253624200
// MI455X (gfx1250) — matrix operand order repaired and checked by reading, not compiled, not run on hardware
//
#include <hip/hip_runtime.h>

typedef __bf16 v16bf __attribute__((ext_vector_type(16)));
typedef float  v8f   __attribute__((ext_vector_type(8)));
typedef int    v4i   __attribute__((ext_vector_type(4)));

typedef __attribute__((address_space(1))) v4i as1_v4i;  // global
typedef __attribute__((address_space(3))) v4i as3_v4i;  // LDS

#ifndef __has_builtin
#define __has_builtin(x) 0
#endif

#if __has_builtin(__builtin_amdgcn_global_load_async_to_lds_b128)
#define HAVE_ASYNC_LDS 1
#else
#define HAVE_ASYNC_LDS 0
#endif

// 16-byte global -> LDS copy. Async (ASYNCcnt-tracked, no VGPR round trip)
// when the gfx1250 builtin is available; plain copy otherwise.
static __device__ __forceinline__ void async_copy16(const void* g, void* l) {
#if HAVE_ASYNC_LDS
  __builtin_amdgcn_global_load_async_to_lds_b128((as1_v4i*)(g), (as3_v4i*)(l),
                                                 0, 0);
#else
  *(uint4*)l = *(const uint4*)g;
#endif
}

static __device__ __forceinline__ void async_join() {
#if HAVE_ASYNC_LDS
#if __has_builtin(__builtin_amdgcn_s_wait_asynccnt)
  __builtin_amdgcn_s_wait_asynccnt(0);
#else
  asm volatile("s_wait_asynccnt 0" ::: "memory");
#endif
#endif
}

union FragB16 { v4i q[2]; v16bf v; };

static __device__ __forceinline__ v8f vzero8() {
  v8f z;
#pragma unroll
  for (int i = 0; i < 8; ++i) z[i] = 0.f;
  return z;
}

static __device__ __forceinline__ v8f wmma_bf16(v16bf a, v16bf b, v8f c) {
  // D = A(16x32 bf16) * B(32x16 bf16) + C(16x16 f32)
  return __builtin_amdgcn_wmma_f32_16x16x32_bf16(false, a, false, b, (short)0, c,
                                                 false, false);
}

// ---- A fragment: 16x32 bf16 tile, LDS row-major [16][stride] ----
// ISA 7.12.2: lane m=lane&15, g=lane>>4; VGPRs 0..3 hold K pairs {0,2,4,6}+g*8
// (bytes [0,16) of row+g*8), VGPRs 4..7 hold {16,18,20,22}+g*8 (bytes [32,48)).
// -> exactly two ds_load_b128 per fragment.
static __device__ __forceinline__ v16bf load_a_frag(const __bf16* tile, int lane,
                                                    int stride, int kofs) {
  const int m = lane & 15, g = lane >> 4;
  const __bf16* row = tile + m * stride + kofs + g * 8;
  FragB16 f;
  f.q[0] = *(const v4i*)(row);
  f.q[1] = *(const v4i*)(row + 16);
  return f.v;
}

// ---- B fragment: 32x16 bf16 (KxN). Lane holds column n = lane&15;
// elements 0..7 -> K = g*8+e, elements 8..15 -> K = 16+g*8+(e-8) -> two ds_load_b128.
// Source tile stored N-major in LDS: tile[n*stride + k].
static __device__ __forceinline__ v16bf load_b_frag(const __bf16* tile, int lane,
                                                    int stride, int kofs) {
  const int n = lane & 15, g = lane >> 4;
  const __bf16* p = tile + n * stride + kofs + g * 8;
  FragB16 f;
  f.q[0] = *(const v4i*)(p);
  f.q[1] = *(const v4i*)(p + 16);
  return f.v;
}

// =====================================================================
// One-shot fp32 -> bf16 convert (for inputs/weights).
// =====================================================================
__global__ __launch_bounds__(256) void cvt_f32_to_bf16(
    const float* __restrict__ src, __bf16* __restrict__ dst, int n4) {
  const int idx = blockIdx.x * 256 + threadIdx.x;
  if (idx >= n4) return;
  float4 v = ((const float4*)src)[idx];
  __bf16* d = dst + (size_t)idx * 4;
  d[0] = (__bf16)v.x; d[1] = (__bf16)v.y;
  d[2] = (__bf16)v.z; d[3] = (__bf16)v.w;
}

// =====================================================================
// Generic GEMM: C[M,N] = A[M,K] @ W[N,K]^T (+bias) (+residual), A/W bf16.
// Optional outputs: fp32 C, bf16 Cbf, bf16 transposed CbfT[N,M].
// Block 256 threads (8 waves), tile 128x128, async LDS staging.
// LDS rows padded 32->40 bf16 (80 B = 20 dwords, gcd(20,64)=4 -> 16 distinct
// banks across the 16 lanes of a fragment load).
// =====================================================================
#define GK 40  // padded chunk stride (elements)
template <bool BIAS, bool RES, bool OUTF, bool OUTB, bool OUTT>
__global__ __launch_bounds__(256) void gemm_bf16_wmma(
    const __bf16* __restrict__ A, const __bf16* __restrict__ W,
    const float* __restrict__ bias, const float* __restrict__ residual,
    float* __restrict__ C, __bf16* __restrict__ Cbf, __bf16* __restrict__ CbfT,
    int M, int N, int K) {
  __shared__ __bf16 a_lds[128 * GK];
  __shared__ __bf16 b_lds[128 * GK];
  const int tid = threadIdx.x;
  const int lane = tid & 31, w = tid >> 5;
  const int m0 = blockIdx.x * 128, n0 = blockIdx.y * 128;
  const int row = tid >> 1, half = tid & 1;

  v8f acc[8];
#pragma unroll
  for (int t = 0; t < 8; ++t) acc[t] = vzero8();

  for (int kc = 0; kc < K; kc += 32) {
    __syncthreads();
    {
      const __bf16* Ap = A + (size_t)(m0 + row) * K + kc + half * 16;
      const __bf16* Wp = W + (size_t)(n0 + row) * K + kc + half * 16;
      __bf16* ad = &a_lds[row * GK + half * 16];
      __bf16* bd = &b_lds[row * GK + half * 16];
      async_copy16(Ap, ad);
      async_copy16(Ap + 8, ad + 8);
      async_copy16(Wp, bd);
      async_copy16(Wp + 8, bd + 8);
    }
    async_join();
    __syncthreads();
    v16bf a = load_a_frag(a_lds + w * 16 * GK, lane, GK, 0);
#pragma unroll
    for (int t = 0; t < 8; ++t) {
      v16bf b = load_b_frag(b_lds + t * 16 * GK, lane, GK, 0);
      acc[t] = wmma_bf16(a, b, acc[t]);
    }
  }

  // C/D layout: VGPR r -> row m0+w*16+g*8+r, col n0+t*16+n
  const int n = lane & 15, g = lane >> 4;
#pragma unroll
  for (int t = 0; t < 8; ++t) {
#pragma unroll
    for (int r = 0; r < 8; ++r) {
      const int rr = m0 + w * 16 + g * 8 + r;
      const int cc = n0 + t * 16 + n;
      float v = acc[t][r];
      if constexpr (BIAS) v += bias[cc];
      if constexpr (RES) v += residual[(size_t)rr * N + cc];
      if constexpr (OUTF) C[(size_t)rr * N + cc] = v;
      if constexpr (OUTB) Cbf[(size_t)rr * N + cc] = (__bf16)v;
      if constexpr (OUTT) CbfT[(size_t)cc * M + rr] = (__bf16)v;
    }
  }
}

// =====================================================================
// Per-(batch,head) softmax attention: S=128, HD=16. qkv fp32 [B*S, 768].
// Writes bf16 ctx[B*S, 256] with heads re-interleaved (h*16+d).
// =====================================================================
__global__ __launch_bounds__(128) void attn_softmax_kernel(
    const float* __restrict__ qkv, __bf16* __restrict__ ctx_bf) {
  extern __shared__ float sm[];
  float* Ks = sm;          // 128*16
  float* Vs = sm + 2048;   // 128*16
  float* Sc = sm + 4096;   // 128*128
  const int b = blockIdx.x, h = blockIdx.y;
  const int t = threadIdx.x;  // query index 0..127
  const float* base = qkv + (size_t)b * 128 * 768;

  float q[16];
#pragma unroll
  for (int d = 0; d < 16; ++d) {
    q[d] = base[(size_t)t * 768 + h * 16 + d];
    Ks[t * 16 + d] = base[(size_t)t * 768 + 256 + h * 16 + d];
    Vs[t * 16 + d] = base[(size_t)t * 768 + 512 + h * 16 + d];
  }
  __syncthreads();

  float mx = -3.402823466e38f;
  for (int k = 0; k < 128; ++k) {
    float s = 0.f;
#pragma unroll
    for (int d = 0; d < 16; ++d) s += q[d] * Ks[k * 16 + d];
    s *= 0.25f;  // 1/sqrt(HD=16)
    Sc[t * 128 + k] = s;
    mx = fmaxf(mx, s);
  }
  float sum = 0.f;
  for (int k = 0; k < 128; ++k) {
    float e = __expf(Sc[t * 128 + k] - mx);
    Sc[t * 128 + k] = e;
    sum += e;
  }
  const float inv = 1.f / sum;
#pragma unroll
  for (int d = 0; d < 16; ++d) {
    float o = 0.f;
    for (int k = 0; k < 128; ++k) o += Sc[t * 128 + k] * Vs[k * 16 + d];
    ctx_bf[((size_t)(b * 128 + t)) * 256 + h * 16 + d] = (__bf16)(o * inv);
  }
}

// =====================================================================
// Fused: agg[i,:] = (user[i,:] @ seq^T) @ seq / (max(||.||_1,eps) * M)
// All operands pre-converted bf16; staging via async global->LDS copies.
// Block: 128 user rows, 256 threads (8 waves). O tiles (16/wave) live in
// registers across all 64 seq chunks; per-wave LDS scratch reshapes S into
// A-fragments (same-wave DS ops are in-order -> no barrier needed).
// Row strides padded (+8 bf16) for bank-conflict-free b128 fragment loads.
// =====================================================================
#define S256 264  // padded stride for 256-wide tiles
#define S128 136  // padded stride for 128-wide tiles
__global__ __launch_bounds__(256) void fused_att_agg(
    const __bf16* __restrict__ user_bf, const __bf16* __restrict__ seq_bf,
    const __bf16* __restrict__ seqT_bf, __bf16* __restrict__ agg_bf) {
  extern __shared__ char smem[];
  __bf16* u_lds  = (__bf16*)smem;          // 128 x S256 user tile
  __bf16* sq_lds = u_lds + 128 * S256;     // 128 x S256 seq chunk row-major
  __bf16* st_lds = sq_lds + 128 * S256;    // 256 x S128 seq chunk transposed
  __bf16* s_scr  = st_lds + 256 * S128;    // 8 waves * 16 x S128 S scratch
  __shared__ float rowsum[128];

  const int tid = threadIdx.x;
  const int lane = tid & 31, w = tid >> 5;
  const int n = lane & 15, g = lane >> 4;
  const int i0 = blockIdx.x * 128;
  const int row = tid >> 1, half = tid & 1;

  if (tid < 128) rowsum[tid] = 0.f;

  // Stage user tile (async; joined by the first chunk's async_join)
  {
    const __bf16* Up = user_bf + (size_t)(i0 + row) * 256 + half * 128;
    __bf16* Ud = &u_lds[row * S256 + half * 128];
#pragma unroll
    for (int i = 0; i < 16; ++i) async_copy16(Up + i * 8, Ud + i * 8);
  }

  v8f o_acc[16];
#pragma unroll
  for (int ct = 0; ct < 16; ++ct) o_acc[ct] = vzero8();

  for (int jc = 0; jc < 8192; jc += 128) {
    __syncthreads();
    // Stage seq chunk: row-major copy + pre-transposed copy, both async
    {
      const __bf16* Sp = seq_bf + (size_t)(jc + row) * 256 + half * 128;
      __bf16* Sd = &sq_lds[row * S256 + half * 128];
#pragma unroll
      for (int i = 0; i < 16; ++i) async_copy16(Sp + i * 8, Sd + i * 8);
      const __bf16* Tp = seqT_bf + (size_t)tid * 8192 + jc;
      __bf16* Td = &st_lds[tid * S128];
#pragma unroll
      for (int i = 0; i < 16; ++i) async_copy16(Tp + i * 8, Td + i * 8);
    }
    async_join();
    __syncthreads();

    // ---- S = U_tile @ seq_chunk^T : wave w owns rows w*16..+15, 128 cols ----
    v8f s_acc[8];
#pragma unroll
    for (int t = 0; t < 8; ++t) s_acc[t] = vzero8();
#pragma unroll
    for (int kk = 0; kk < 256; kk += 32) {
      v16bf a = load_a_frag(u_lds + w * 16 * S256, lane, S256, kk);
#pragma unroll
      for (int t = 0; t < 8; ++t) {
        v16bf b = load_b_frag(sq_lds + t * 16 * S256, lane, S256, kk);
        s_acc[t] = wmma_bf16(a, b, s_acc[t]);
      }
    }

    // L1 row partial sums (ds_add_f32) + S -> bf16 per-wave scratch
#pragma unroll
    for (int r = 0; r < 8; ++r) {
      float p = 0.f;
#pragma unroll
      for (int t = 0; t < 8; ++t) p += fabsf(s_acc[t][r]);
      atomicAdd(&rowsum[w * 16 + g * 8 + r], p);
    }
#pragma unroll
    for (int t = 0; t < 8; ++t)
#pragma unroll
      for (int r = 0; r < 8; ++r)
        s_scr[w * 16 * S128 + (g * 8 + r) * S128 + t * 16 + n] =
            (__bf16)s_acc[t][r];

    // ---- O += S @ seq_chunk : wave w rows w*16..+15, 256 cols ----
#pragma unroll
    for (int t4 = 0; t4 < 4; ++t4) {
      v16bf a = load_a_frag(s_scr + w * 16 * S128, lane, S128, t4 * 32);
#pragma unroll
      for (int ct = 0; ct < 16; ++ct) {
        v16bf b = load_b_frag(st_lds + ct * 16 * S128, lane, S128, t4 * 32);
        o_acc[ct] = wmma_bf16(a, b, o_acc[ct]);
      }
    }
  }
  __syncthreads();  // rowsum complete

#pragma unroll
  for (int r = 0; r < 8; ++r) {
    const float rs = rowsum[w * 16 + g * 8 + r];
    const float sc = 1.f / (fmaxf(rs, 1e-12f) * 8192.f);  // /M folded in
    const size_t rowoff = (size_t)(i0 + w * 16 + g * 8 + r) * 256;
#pragma unroll
    for (int ct = 0; ct < 16; ++ct)
      agg_bf[rowoff + ct * 16 + n] = (__bf16)(o_acc[ct][r] * sc);
  }
}

extern "C" void kernel_launch(void* const* d_in, const int* in_sizes, int n_in,
                              void* d_out, int out_size, void* d_ws, size_t ws_size,
                              hipStream_t stream) {
  const float* user = (const float*)d_in[0];  // [16384, 256]
  const float* seqf = (const float*)d_in[1];  // [64, 128, 256]
  const float* ipw  = (const float*)d_in[2];  // [768, 256]
  const float* ipb  = (const float*)d_in[3];  // [768]
  const float* opw  = (const float*)d_in[4];  // [256, 256]
  const float* opb  = (const float*)d_in[5];  // [256]
  const float* tw   = (const float*)d_in[6];  // [256, 256]
  const float* tb   = (const float*)d_in[7];  // [256]
  float* out = (float*)d_out;                 // [16384, 256]

  // Workspace: fp32 qkv, then bf16 buffers
  float* qkv = (float*)d_ws;                           // 8192*768 f32
  __bf16* bf = (__bf16*)(qkv + (size_t)8192 * 768);
  __bf16* seqf_bf = bf; bf += (size_t)8192 * 256;
  __bf16* ipw_bf  = bf; bf += (size_t)768 * 256;
  __bf16* opw_bf  = bf; bf += (size_t)256 * 256;
  __bf16* tw_bf   = bf; bf += (size_t)256 * 256;
  __bf16* user_bf = bf; bf += (size_t)16384 * 256;
  __bf16* ctx_bf  = bf; bf += (size_t)8192 * 256;
  __bf16* seq_bf  = bf; bf += (size_t)8192 * 256;
  __bf16* seqT_bf = bf; bf += (size_t)256 * 8192;
  __bf16* agg_bf  = bf;

  // 0) one-shot fp32 -> bf16 conversions
  auto cvt = [&](const float* s, __bf16* d, size_t elems) {
    const int n4 = (int)(elems / 4);
    cvt_f32_to_bf16<<<dim3((n4 + 255) / 256), dim3(256), 0, stream>>>(s, d, n4);
  };
  cvt(seqf, seqf_bf, (size_t)8192 * 256);
  cvt(ipw,  ipw_bf,  (size_t)768 * 256);
  cvt(opw,  opw_bf,  (size_t)256 * 256);
  cvt(tw,   tw_bf,   (size_t)256 * 256);
  cvt(user, user_bf, (size_t)16384 * 256);

  const dim3 blk(256);
  // 1) qkv = seq_feat @ in_proj_w^T + in_proj_b   [8192 x 768] fp32
  gemm_bf16_wmma<true, false, true, false, false><<<dim3(64, 6), blk, 0, stream>>>(
      seqf_bf, ipw_bf, ipb, nullptr, qkv, nullptr, nullptr, 8192, 768, 256);
  // 2) per-head softmax attention -> ctx (bf16)
  attn_softmax_kernel<<<dim3(64, 16), dim3(128), 81920, stream>>>(qkv, ctx_bf);
  // 3) seq = ctx @ out_proj_w^T + out_proj_b -> bf16 row-major AND transposed
  gemm_bf16_wmma<true, false, false, true, true><<<dim3(64, 2), blk, 0, stream>>>(
      ctx_bf, opw_bf, opb, nullptr, nullptr, seq_bf, seqT_bf, 8192, 256, 256);
  // 4) fused att / L1-normalize / aggregate -> agg (bf16)
  //    LDS: (128*264 + 128*264 + 256*136 + 8*16*136) * 2 = 239616 bytes
  fused_att_agg<<<dim3(128), blk, 239616, stream>>>(user_bf, seq_bf, seqT_bf,
                                                    agg_bf);
  // 5) out = user + agg @ trans_w^T + trans_b     [16384 x 256] fp32
  gemm_bf16_wmma<true, true, true, false, false><<<dim3(128, 2), blk, 0, stream>>>(
      agg_bf, tw_bf, tb, user, out, nullptr, nullptr, 16384, 256, 256);
}
